// MyModel_25658134627023
// MI455X (gfx1250) — matrix operand order repaired and checked by reading, not compiled, not run on hardware
//
#include <hip/hip_runtime.h>
#include <stdint.h>

#ifndef __has_builtin
#define __has_builtin(x) 0
#endif

#if defined(__AMDGCN__) && __has_builtin(__builtin_amdgcn_tensor_load_to_lds)
#define HAVE_TDM 1
#else
#define HAVE_TDM 0
#endif

// ---------------- problem dims ----------------
#define N_   1024
#define L_   128
#define D_   300
#define H_   150
#define G_   600      // 4*H
#define GP   640      // G padded so GP/16 divides evenly over 8 waves
#define NT   40       // GP/16 column tiles
#define TPW  5        // tiles per wave (NT / 8 waves) -> compile-time unroll, regs not scratch
#define KC1  480      // concat K for LSTM1: x pad 320 + h pad 160
#define KT1  15       // KC1/32 k-tiles (bf16 wmma K=32)
#define KC2  320      // K for LSTM2 input projection (D padded)
#define KT2  10

typedef __bf16 bf16_t;
typedef bf16_t v16bf __attribute__((ext_vector_type(16)));
typedef float  v8f   __attribute__((ext_vector_type(8)));
typedef unsigned int u32x4 __attribute__((ext_vector_type(4)));
typedef int          i32x4 __attribute__((ext_vector_type(4)));
typedef int          i32x8 __attribute__((ext_vector_type(8)));

__device__ __forceinline__ unsigned short f2bf(float f) {
  unsigned int u = __float_as_uint(f);
  u += 0x7FFFu + ((u >> 16) & 1u);          // round-to-nearest-even
  return (unsigned short)(u >> 16);
}
__device__ __forceinline__ float bf2f(unsigned short h) {
  return __uint_as_float(((unsigned int)h) << 16);
}
__device__ __forceinline__ float sigm(float x) { return 1.f / (1.f + __expf(-x)); }

__device__ __forceinline__ void tensor_wait0() {
#if defined(__AMDGCN__)
#if __has_builtin(__builtin_amdgcn_s_wait_tensorcnt)
  __builtin_amdgcn_s_wait_tensorcnt((short)0);
#else
  asm volatile("s_wait_tensorcnt 0x0" ::: "memory");
#endif
#endif
}

#if HAVE_TDM
// 2D TDM tile load: rows x (16 dwords) from global (row stride strideDW dwords)
// into LDS with 1-dword pad after every 16 dwords (-> LDS row stride 17 dwords).
__device__ __forceinline__ void tdm_load_slab(const unsigned int* gbase, unsigned ldsByteAddr,
                                              int rows, int strideDW) {
  unsigned long long ga = (unsigned long long)(uintptr_t)gbase;
  u32x4 g0 = { 1u,                                   // count=1, no gather
               ldsByteAddr,                          // lds_addr
               (unsigned)ga,                         // global_addr[31:0]
               (unsigned)(ga >> 32) | 0x80000000u }; // global_addr[56:32] | type=2
  // group1: data_size=4B(2), pad_enable, pad_interval=16dw(3), pad_amount=1dw(0)
  int d0 = (int)((2u << 16) | (1u << 20) | (3u << 22));
  int d1 = (int)(((unsigned)strideDW & 0xFFFFu) << 16);          // tensor_dim0 lo16
  int d2 = (int)(((unsigned)strideDW >> 16) |
                 (((unsigned)rows & 0xFFFFu) << 16));            // dim0 hi | tensor_dim1 lo16
  int d3 = (int)(((unsigned)rows >> 16) | (16u << 16));          // dim1 hi | tile_dim0=16
  int d4 = (int)((unsigned)rows & 0xFFFFu);                      // tile_dim1 = rows
  int d5 = (int)(unsigned)strideDW;                              // tensor_dim0_stride lo32
  i32x8 g1 = { d0, d1, d2, d3, d4, d5, 0, 0 };
  i32x4 g2 = { 0, 0, 0, 0 };
  i32x4 g3 = { 0, 0, 0, 0 };
#if __clang_major__ >= 23
  i32x8 g4 = { 0, 0, 0, 0, 0, 0, 0, 0 };
  __builtin_amdgcn_tensor_load_to_lds(g0, g1, g2, g3, g4, 0);
#else
  __builtin_amdgcn_tensor_load_to_lds(g0, g1, g2, g3, 0);
#endif
}
#endif

// ---------------- weight prep: pack [W_ih | pad | W_hh | pad] as bf16 [GP][Kpad] ----------------
__global__ void prep_wcat(unsigned short* dst, const float* wih, const float* whh, int Kpad) {
  int i = blockIdx.x * blockDim.x + threadIdx.x;
  int total = GP * Kpad;
  if (i >= total) return;
  int col = i / Kpad, k = i % Kpad;
  float v = 0.f;
  if (col < G_) {
    if (k < D_)                                          v = wih[col * D_ + k];
    else if (whh != nullptr && k >= 320 && k < 320 + H_) v = whh[col * H_ + (k - 320)];
  }
  dst[i] = f2bf(v);
}

__global__ void init_state(float* hf, float* cf, float* hmf,
                           float* hb, float* cb, float* hmb) {
  int i = blockIdx.x * blockDim.x + threadIdx.x;
  if (i >= N_ * H_) return;
  hf[i] = 0.f; cf[i] = 0.f; hb[i] = 0.f; cb[i] = 0.f;
  hmf[i] = -3.0e38f; hmb[i] = -3.0e38f;
}

__device__ __forceinline__ float fetchA1(const float* x, const float* hbuf, int n, int xr, int k) {
  if (k < D_) return x[((size_t)n * L_ + xr) * D_ + k];
  if (k >= 320 && k < 320 + H_) return hbuf[n * H_ + (k - 320)];
  return 0.f;
}

// ---------------- LSTM1 one timestep, both directions, WMMA gates + fused gating/pool ----------------
__global__ __launch_bounds__(256) void lstm1_step(
    const float* __restrict__ x, const int* __restrict__ lengths,
    const unsigned short* __restrict__ wcat_f, const unsigned short* __restrict__ wcat_b,
    const float* __restrict__ bias_f, const float* __restrict__ bias_b,
    float* __restrict__ hf, float* __restrict__ cf, float* __restrict__ hmf,
    float* __restrict__ hb, float* __restrict__ cb, float* __restrict__ hmb,
    int t)
{
  __shared__ unsigned int Alds[16 * 241];   // 16 rows x 240 bf16-pairs, stride 241 (conflict-free)
  __shared__ unsigned int Blds[GP * 17];    // 640 cols x 16 dwords (one K=32 slab), stride 17
  __shared__ float        Glds[16 * GP];    // gates tile

  const int tid  = threadIdx.x;
  const int lane = tid & 31;
  const int wv   = tid >> 5;
  const int mtile = blockIdx.x;             // 0..63
  const int dir   = blockIdx.y;             // 0 fwd, 1 bwd

  const unsigned int* wcatU = (const unsigned int*)(dir ? wcat_b : wcat_f);
  const float* bias = dir ? bias_b : bias_f;
  float* hbuf = dir ? hb : hf;
  float* cbuf = dir ? cb : cf;
  float* hmax = dir ? hmb : hmf;

  // stage A tile: 16 rows, K = [x_t(300)|0(20)|h(150)|0(10)] as packed bf16 pairs
  for (int p = tid; p < 16 * 240; p += 256) {
    int m = p / 240, kp = p % 240, k = kp * 2;
    int n = mtile * 16 + m;
    int ln = lengths[n];
    int xr = dir ? ((t < ln) ? (ln - 1 - t) : t) : t;   // ragged reverse for bwd
    float v0 = fetchA1(x, hbuf, n, xr, k);
    float v1 = fetchA1(x, hbuf, n, xr, k + 1);
    Alds[m * 241 + kp] = (unsigned int)f2bf(v0) | ((unsigned int)f2bf(v1) << 16);
  }

  v8f acc[TPW];
  const v8f vz = {0.f, 0.f, 0.f, 0.f, 0.f, 0.f, 0.f, 0.f};
#pragma unroll
  for (int i = 0; i < TPW; ++i) acc[i] = vz;

  for (int kk = 0; kk < KT1; ++kk) {
    __syncthreads();                       // previous slab fully consumed
#if HAVE_TDM
    if (wv == 0) {                         // one wave drives the Tensor Data Mover
      tdm_load_slab(wcatU + (size_t)kk * 16, (unsigned)(uintptr_t)&Blds[0],
                    GP, KC1 / 2);
      tensor_wait0();
    }
#else
    for (int i = tid; i < GP * 16; i += 256) {
      int c = i >> 4, q = i & 15;
      Blds[c * 17 + q] = wcatU[(((size_t)c * KC1 + (size_t)kk * 32) >> 1) + q];
    }
#endif
    if (kk + 1 < KT1)
      __builtin_prefetch(&wcatU[(((size_t)(tid << 1) * KC1 + (size_t)(kk + 1) * 32) >> 1)], 0, 0);
    __syncthreads();

    // A fragment (16x32 bf16, ISA 7.12.2 layout)
    union { unsigned int u[8]; v16bf v; } afr;
    const int m = lane & 15;
#pragma unroll
    for (int v = 0; v < 8; ++v) {
      int k = kk * 32 + ((v >> 2) << 4) + ((lane >> 4) << 3) + ((v & 3) << 1);
      afr.u[v] = Alds[m * 241 + (k >> 1)];
    }
#pragma unroll
    for (int ji = 0; ji < TPW; ++ji) {     // compile-time tile index -> acc stays in VGPRs
      int jt = wv + ji * 8;
      union { unsigned int u[8]; v16bf v; } bfr;
      int col = jt * 16 + (lane & 15);
#pragma unroll
      for (int v = 0; v < 8; ++v)          // B 32x16: K = 16*(v>=4)+8*(lane>=16)+2*(v&3)
        bfr.u[v] = Blds[col * 17 + ((v >> 2) << 3) + ((lane >> 4) << 2) + (v & 3)];
      acc[ji] = __builtin_amdgcn_wmma_f32_16x16x32_bf16(
          false, afr.v, false, bfr.v, (short)0, acc[ji], false, false);
    }
  }

  __syncthreads();
#pragma unroll
  for (int ji = 0; ji < TPW; ++ji) {
    int jt = wv + ji * 8;
    int coln = jt * 16 + (lane & 15);
    int rbase = (lane >> 4) << 3;
#pragma unroll
    for (int r = 0; r < 8; ++r)
      Glds[(r + rbase) * GP + coln] = acc[ji][r];
  }
  __syncthreads();

  // fused gating + state update + masked running-max pool
  for (int p = tid; p < 16 * H_; p += 256) {
    int m = p / H_, j = p % H_;
    int n = mtile * 16 + m;
    float gi = Glds[m * GP + j]          + bias[j];
    float gf = Glds[m * GP + H_ + j]     + bias[H_ + j];
    float gg = Glds[m * GP + 2 * H_ + j] + bias[2 * H_ + j];
    float go = Glds[m * GP + 3 * H_ + j] + bias[3 * H_ + j];
    float cold = cbuf[n * H_ + j];
    float cnew = sigm(gf) * cold + sigm(gi) * tanhf(gg);
    float hnew = sigm(go) * tanhf(cnew);
    cbuf[n * H_ + j] = cnew;
    hbuf[n * H_ + j] = hnew;
    if (t < lengths[n]) {
      float* hm = &hmax[n * H_ + j];
      *hm = fmaxf(*hm, hnew);
    }
  }
}

// ---------------- small VALU stages ----------------
__global__ void build_sent(const float* hmf, const float* hmb, float* sent) {
  int i = blockIdx.x * blockDim.x + threadIdx.x;
  if (i >= N_ * D_) return;
  int n = i / D_, d = i % D_;
  sent[i] = (d < H_) ? hmf[n * H_ + d] : hmb[n * H_ + (d - H_)];
}

__global__ void ex_logits(const float* sent, const float* w_ex, const float* b_ex, float* exlog) {
  int i = blockIdx.x * blockDim.x + threadIdx.x;
  if (i >= N_ * 7) return;
  int n = i / 7, c = i % 7;
  float a = b_ex[c];
  const float* s = sent + (size_t)n * D_;
  const float* w = w_ex + (size_t)c * D_;
  for (int d = 0; d < D_; ++d) a += s[d] * w[d];
  exlog[i] = a;
}

__global__ void ex_softmax(const float* exlog, float* exout, int* expre) {
  int n = blockIdx.x * blockDim.x + threadIdx.x;
  if (n >= N_) return;
  float l[7]; float mx = -1e30f; int am = 0;
  for (int c = 0; c < 7; ++c) { l[c] = exlog[n * 7 + c]; if (l[c] > mx) { mx = l[c]; am = c; } }
  float s = 0.f;
  for (int c = 0; c < 7; ++c) { l[c] = __expf(l[c] - mx); s += l[c]; }
  for (int c = 0; c < 7; ++c) exout[n * 7 + c] = l[c] / s;
  expre[n] = am;
}

__global__ void avg_lab_k(const float* exout, const float* lemb, float* avglab) {
  int i = blockIdx.x * blockDim.x + threadIdx.x;
  if (i >= N_ * D_) return;
  int n = i / D_, d = i % D_;
  float a = 0.f;
  for (int c = 0; c < 7; ++c) a += exout[n * 7 + c] * lemb[c * D_ + d];
  avglab[i] = a;
}

__global__ void joint_mlp(const float* sent, const float* avglab,
                          const float* w_mlp, const float* b_mlp, float* jointv) {
  int i = blockIdx.x * blockDim.x + threadIdx.x;
  if (i >= N_ * D_) return;
  int n = i / D_, o = i % D_;
  float a = b_mlp[o];
  const float* w = w_mlp + (size_t)o * (2 * D_);
  const float* s = sent + (size_t)n * D_;
  const float* v = avglab + (size_t)n * D_;
  for (int k = 0; k < D_; ++k) a += s[k] * w[k];
  for (int k = 0; k < D_; ++k) a += v[k] * w[D_ + k];
  jointv[i] = fmaxf(a, 0.f);
}

__global__ void score_raw_k(const float* jointv, const float* w_sc, const float* b_sc, float* sraw) {
  int n = blockIdx.x * blockDim.x + threadIdx.x;
  if (n >= N_) return;
  float a = b_sc[0];
  const float* j = jointv + (size_t)n * D_;
  for (int d = 0; d < D_; ++d) a += j[d] * w_sc[d];
  sraw[n] = sigm(a);
}

__global__ __launch_bounds__(1024) void score_softmax(const float* sraw, float* score) {
  __shared__ float red[1024];
  int tid = threadIdx.x;
  float v = sraw[tid];
  red[tid] = v; __syncthreads();
  for (int o = 512; o > 0; o >>= 1) { if (tid < o) red[tid] = fmaxf(red[tid], red[tid + o]); __syncthreads(); }
  float mx = red[0]; __syncthreads();
  float e = __expf(v - mx);
  red[tid] = e; __syncthreads();
  for (int o = 512; o > 0; o >>= 1) { if (tid < o) red[tid] += red[tid + o]; __syncthreads(); }
  score[tid] = e / red[0];
}

__global__ __launch_bounds__(1024) void topk_select(const float* score, int* intop) {
  __shared__ float sc[1024];
  int tid = threadIdx.x;
  float v = score[tid];
  sc[tid] = v; __syncthreads();
  int cnt = 0;
  for (int m = 0; m < 1024; ++m) {
    float u = sc[m];
    cnt += ((u > v) || (u == v && m < tid)) ? 1 : 0;   // top_k tie-break: lower index first
  }
  intop[tid] = (cnt < N_ / 2) ? 1 : 0;
}

__global__ void mixed_bf_k(const float* jointv, const float* sent, const int* intop,
                           unsigned short* mbf) {
  int i = blockIdx.x * blockDim.x + threadIdx.x;
  if (i >= N_ * KC2) return;
  int n = i / KC2, k = i % KC2;
  float v = 0.f;
  if (k < D_) v = intop[n] ? jointv[n * D_ + k] : sent[n * D_ + k];
  mbf[i] = f2bf(v);
}

// ---------------- LSTM2 input projection GEMM (WMMA) ----------------
__global__ __launch_bounds__(256) void proj2_wmma(
    const unsigned short* __restrict__ abf,
    const unsigned short* __restrict__ w2fb, const unsigned short* __restrict__ w2bb,
    const float* __restrict__ b2f_, const float* __restrict__ b2b_,
    float* __restrict__ outf, float* __restrict__ outb)
{
  const int tid = threadIdx.x, lane = tid & 31, wv = tid >> 5;
  const int mtile = blockIdx.x, dir = blockIdx.y;
  const unsigned int* AU = (const unsigned int*)abf;
  const unsigned int* WU = (const unsigned int*)(dir ? w2bb : w2fb);
  const float* bias = dir ? b2b_ : b2f_;
  float* out = dir ? outb : outf;

  v8f acc[TPW];
  const v8f vz = {0.f, 0.f, 0.f, 0.f, 0.f, 0.f, 0.f, 0.f};
#pragma unroll
  for (int i = 0; i < TPW; ++i) acc[i] = vz;

  for (int kk = 0; kk < KT2; ++kk) {
    union { unsigned int u[8]; v16bf v; } afr;
    int row = mtile * 16 + (lane & 15);
#pragma unroll
    for (int v = 0; v < 8; ++v) {
      int k = kk * 32 + ((v >> 2) << 4) + ((lane >> 4) << 3) + ((v & 3) << 1);
      afr.u[v] = AU[((size_t)row * KC2 + k) >> 1];
    }
#pragma unroll
    for (int ji = 0; ji < TPW; ++ji) {
      int jt = wv + ji * 8;
      union { unsigned int u[8]; v16bf v; } bfr;
      int col = jt * 16 + (lane & 15);
#pragma unroll
      for (int v = 0; v < 8; ++v) {
        int kl = kk * 32 + ((v >> 2) << 4) + ((lane >> 4) << 3) + ((v & 3) << 1);
        bfr.u[v] = WU[((size_t)col * KC2 + kl) >> 1];
      }
      acc[ji] = __builtin_amdgcn_wmma_f32_16x16x32_bf16(
          false, afr.v, false, bfr.v, (short)0, acc[ji], false, false);
    }
  }
#pragma unroll
  for (int ji = 0; ji < TPW; ++ji) {
    int jt = wv + ji * 8;
    int col = jt * 16 + (lane & 15);
    if (col < G_) {
      int rbase = (lane >> 4) << 3;
#pragma unroll
      for (int r = 0; r < 8; ++r) {
        int row = mtile * 16 + rbase + r;
        out[(size_t)row * G_ + col] = acc[ji][r] + bias[col];
      }
    }
  }
}

// ---------------- LSTM2: persistent sequential recurrence, W_hh in LDS as bf16 ----------------
__global__ __launch_bounds__(256) void lstm2_seq(
    const float* __restrict__ projf, const float* __restrict__ projb,
    const float* __restrict__ whhf, const float* __restrict__ whhb,
    float* __restrict__ h2f, float* __restrict__ h2b)
{
  __shared__ unsigned short Wl[G_ * 152];   // 600x150 bf16, row stride 152
  __shared__ float hl[H_], cl[H_], gl[G_];
  const int tid = threadIdx.x;
  const int dir = blockIdx.x;
  const float* proj = dir ? projb : projf;
  const float* whh = dir ? whhb : whhf;
  float* hout = dir ? h2b : h2f;

  for (int i = tid; i < G_ * H_; i += 256) {
    int c = i / H_, j = i % H_;
    Wl[c * 152 + j] = f2bf(whh[i]);
  }
  if (tid < H_) { hl[tid] = 0.f; cl[tid] = 0.f; }
  __syncthreads();

  for (int s = 0; s < N_; ++s) {
    int t = dir ? (N_ - 1 - s) : s;
    for (int col = tid; col < G_; col += 256) {
      float a = proj[(size_t)t * G_ + col];     // bias already folded in
      const unsigned short* wr = &Wl[col * 152];
      for (int j = 0; j < H_; ++j) a += hl[j] * bf2f(wr[j]);
      gl[col] = a;
    }
    __syncthreads();
    if (tid < H_) {
      float gi = gl[tid], gf = gl[H_ + tid], gg = gl[2 * H_ + tid], go = gl[3 * H_ + tid];
      float cn = sigm(gf) * cl[tid] + sigm(gi) * tanhf(gg);
      float hn = sigm(go) * tanhf(cn);
      cl[tid] = cn; hl[tid] = hn;
      hout[(size_t)t * H_ + tid] = hn;
    }
    __syncthreads();
  }
}

__global__ void tag_logits(const float* h2f, const float* h2b,
                           const float* w_tag, const float* b_tag, float* taglog) {
  int i = blockIdx.x * blockDim.x + threadIdx.x;
  if (i >= N_ * 7) return;
  int n = i / 7, c = i % 7;
  float a = b_tag[c];
  const float* w = w_tag + (size_t)c * D_;
  for (int d = 0; d < H_; ++d) a += h2f[n * H_ + d] * w[d];
  for (int d = 0; d < H_; ++d) a += h2b[n * H_ + d] * w[H_ + d];
  taglog[i] = a;
}

__global__ void zero_loss(float* out) { out[N_] = 0.f; }

__global__ __launch_bounds__(256) void finalize(
    const float* taglog, const int* gold, const int* expre,
    const int* intop, const float* score, float* out)
{
  int n = blockIdx.x * blockDim.x + threadIdx.x;
  if (n >= N_) return;
  float l[7]; float mx = -1e30f; int am = 0;
  for (int c = 0; c < 7; ++c) { l[c] = taglog[n * 7 + c]; if (l[c] > mx) { mx = l[c]; am = c; } }
  float s = 0.f;
  for (int c = 0; c < 7; ++c) s += __expf(l[c] - mx);
  float lse = mx + __logf(s);
  out[n] = (float)am;
  int g = gold[n];
  float ce = -(l[g] - lse);
  int wrong = (g != expre[n]) ? 1 : 0;
  int it = intop[n];
  float pen = 0.f;
  if (wrong && it) pen = 1.5f * score[n];
  else if (!wrong && !it) pen = -1.5f * score[n];
  atomicAdd(&out[N_], ce + pen);
}

// ---------------- host launch ----------------
extern "C" void kernel_launch(void* const* d_in, const int* in_sizes, int n_in,
                              void* d_out, int out_size, void* d_ws, size_t ws_size,
                              hipStream_t stream) {
  (void)in_sizes; (void)n_in; (void)out_size; (void)ws_size;
  const float* x      = (const float*)d_in[0];
  const int* lengths  = (const int*)d_in[1];
  const int* gold     = (const int*)d_in[2];
  const float* w1f_ih = (const float*)d_in[3];
  const float* w1f_hh = (const float*)d_in[4];
  const float* b1f    = (const float*)d_in[5];
  const float* w1b_ih = (const float*)d_in[6];
  const float* w1b_hh = (const float*)d_in[7];
  const float* b1b    = (const float*)d_in[8];
  const float* lemb   = (const float*)d_in[9];
  const float* w_ex   = (const float*)d_in[10];
  const float* b_ex   = (const float*)d_in[11];
  const float* w_mlp  = (const float*)d_in[12];
  const float* b_mlp  = (const float*)d_in[13];
  const float* w_sc   = (const float*)d_in[14];
  const float* b_sc   = (const float*)d_in[15];
  const float* w2f_ih = (const float*)d_in[16];
  const float* w2f_hh = (const float*)d_in[17];
  const float* b2f    = (const float*)d_in[18];
  const float* w2b_ih = (const float*)d_in[19];
  const float* w2b_hh = (const float*)d_in[20];
  const float* b2b    = (const float*)d_in[21];
  const float* w_tag  = (const float*)d_in[22];
  const float* b_tag  = (const float*)d_in[23];
  float* out = (float*)d_out;

  // bump-allocate workspace (~16 MB total)
  char* p = (char*)d_ws;
  auto take = [&](size_t bytes) -> void* {
    void* r = (void*)p;
    p += (bytes + 255) & ~(size_t)255;
    return r;
  };
  unsigned short* wcat_f = (unsigned short*)take((size_t)GP * KC1 * 2);
  unsigned short* wcat_b = (unsigned short*)take((size_t)GP * KC1 * 2);
  unsigned short* w2f_bf = (unsigned short*)take((size_t)GP * KC2 * 2);
  unsigned short* w2b_bf = (unsigned short*)take((size_t)GP * KC2 * 2);
  float* hf   = (float*)take((size_t)N_ * H_ * 4);
  float* cf   = (float*)take((size_t)N_ * H_ * 4);
  float* hmf  = (float*)take((size_t)N_ * H_ * 4);
  float* hb   = (float*)take((size_t)N_ * H_ * 4);
  float* cb   = (float*)take((size_t)N_ * H_ * 4);
  float* hmb  = (float*)take((size_t)N_ * H_ * 4);
  float* sent = (float*)take((size_t)N_ * D_ * 4);
  float* exlog  = (float*)take((size_t)N_ * 7 * 4);
  float* exout  = (float*)take((size_t)N_ * 7 * 4);
  int*   expre  = (int*)take((size_t)N_ * 4);
  float* avglab = (float*)take((size_t)N_ * D_ * 4);
  float* jointv = (float*)take((size_t)N_ * D_ * 4);
  float* sraw   = (float*)take((size_t)N_ * 4);
  float* score  = (float*)take((size_t)N_ * 4);
  int*   intop  = (int*)take((size_t)N_ * 4);
  unsigned short* mixedbf = (unsigned short*)take((size_t)N_ * KC2 * 2);
  float* proj2f = (float*)take((size_t)N_ * G_ * 4);
  float* proj2b = (float*)take((size_t)N_ * G_ * 4);
  float* h2f    = (float*)take((size_t)N_ * H_ * 4);
  float* h2b    = (float*)take((size_t)N_ * H_ * 4);
  float* taglog = (float*)take((size_t)N_ * 7 * 4);

  prep_wcat<<<(GP * KC1 + 255) / 256, 256, 0, stream>>>(wcat_f, w1f_ih, w1f_hh, KC1);
  prep_wcat<<<(GP * KC1 + 255) / 256, 256, 0, stream>>>(wcat_b, w1b_ih, w1b_hh, KC1);
  prep_wcat<<<(GP * KC2 + 255) / 256, 256, 0, stream>>>(w2f_bf, w2f_ih, nullptr, KC2);
  prep_wcat<<<(GP * KC2 + 255) / 256, 256, 0, stream>>>(w2b_bf, w2b_ih, nullptr, KC2);
  init_state<<<(N_ * H_ + 255) / 256, 256, 0, stream>>>(hf, cf, hmf, hb, cb, hmb);

  for (int t = 0; t < L_; ++t)
    lstm1_step<<<dim3(N_ / 16, 2), 256, 0, stream>>>(
        x, lengths, wcat_f, wcat_b, b1f, b1b, hf, cf, hmf, hb, cb, hmb, t);

  build_sent<<<(N_ * D_ + 255) / 256, 256, 0, stream>>>(hmf, hmb, sent);
  ex_logits<<<(N_ * 7 + 255) / 256, 256, 0, stream>>>(sent, w_ex, b_ex, exlog);
  ex_softmax<<<(N_ + 255) / 256, 256, 0, stream>>>(exlog, exout, expre);
  avg_lab_k<<<(N_ * D_ + 255) / 256, 256, 0, stream>>>(exout, lemb, avglab);
  joint_mlp<<<(N_ * D_ + 255) / 256, 256, 0, stream>>>(sent, avglab, w_mlp, b_mlp, jointv);
  score_raw_k<<<(N_ + 255) / 256, 256, 0, stream>>>(jointv, w_sc, b_sc, sraw);
  score_softmax<<<1, 1024, 0, stream>>>(sraw, score);
  topk_select<<<1, 1024, 0, stream>>>(score, intop);
  mixed_bf_k<<<(N_ * KC2 + 255) / 256, 256, 0, stream>>>(jointv, sent, intop, mixedbf);
  proj2_wmma<<<dim3(N_ / 16, 2), 256, 0, stream>>>(mixedbf, w2f_bf, w2b_bf, b2f, b2b, proj2f, proj2b);
  lstm2_seq<<<2, 256, 0, stream>>>(proj2f, proj2b, w2f_hh, w2b_hh, h2f, h2b);
  tag_logits<<<(N_ * 7 + 255) / 256, 256, 0, stream>>>(h2f, h2b, w_tag, b_tag, taglog);
  zero_loss<<<1, 1, 0, stream>>>(out);
  finalize<<<(N_ + 255) / 256, 256, 0, stream>>>(taglog, gold, expre, intop, score, out);
}
